// CrossSwitcher_28157805593226
// MI455X (gfx1250) — matrix operand order repaired and checked by reading, not compiled, not run on hardware
//
#include <hip/hip_runtime.h>
#include <hip/hip_bf16.h>

// ---------------- problem constants (match reference) ----------------
#define DIM   2048
#define NH    16
#define HD    128           // head dim
#define BB    2
#define TT    4096
#define MTOK  (BB*TT)       // 8192 tokens
#define KCONV 4
#define CC    64            // delta-rule chunk length

typedef __attribute__((ext_vector_type(16))) _Float16 v16h;
typedef __attribute__((ext_vector_type(8)))  float    v8f;
typedef __attribute__((ext_vector_type(4)))  unsigned int u32x4;
typedef __attribute__((ext_vector_type(8)))  int      i32x8;
typedef __attribute__((ext_vector_type(4)))  int      i32x4;

union Frag16 { v16h v; uint4 q[2]; };

#if __has_builtin(__builtin_amdgcn_tensor_load_to_lds) && __has_builtin(__builtin_amdgcn_s_wait_tensorcnt)
#define HAS_TDM 1
#else
#define HAS_TDM 0
#endif

#if HAS_TDM
// ---- Tensor Data Mover: 2D tile (tile_rows x 32 f16, row stride = row_len) -> LDS
// D# packing per CDNA5 ISA 8.3/8.4: group0 {count, lds_addr, global_addr, type=2},
// group1 {data_size=1(2B), tensor dims, tile dims, tensor_dim0_stride}.
__device__ __forceinline__ void tdm_load_2d(const void* gaddr, unsigned lds_byte_addr,
                                            unsigned tile_rows, unsigned row_len_elems) {
  unsigned long long ga = (unsigned long long)(size_t)gaddr;
  u32x4 g0;
  g0[0] = 1u;                                              // count=1, user descriptor
  g0[1] = lds_byte_addr;                                   // LDS byte address
  g0[2] = (unsigned)ga;                                    // global_addr[31:0]
  g0[3] = (unsigned)((ga >> 32) & 0x01FFFFFFu) | (2u << 30); // addr[56:32] | type=2
  const unsigned td0 = row_len_elems;                      // tensor dim0 (elems)
  const unsigned td1 = tile_rows;                          // tensor dim1
  i32x8 g1;
  g1[0] = (int)(1u << 16);                                 // data_size = 1 (2 bytes)
  g1[1] = (int)((td0 & 0xFFFFu) << 16);                    // tensor_dim0 lo
  g1[2] = (int)((td0 >> 16) | ((td1 & 0xFFFFu) << 16));    // tensor_dim0 hi | dim1 lo
  g1[3] = (int)((td1 >> 16) | (32u << 16));                // dim1 hi | tile_dim0 = 32
  g1[4] = (int)tile_rows;                                  // tile_dim1 ; tile_dim2 = 0
  g1[5] = (int)row_len_elems;                              // tensor_dim0_stride[31:0]
  g1[6] = 0;                                               // stride hi / dim1_stride lo
  g1[7] = 0;
  i32x4 z4 = {};
#if defined(__clang_major__) && (__clang_major__ >= 23)
  i32x8 z8 = {};
  __builtin_amdgcn_tensor_load_to_lds(g0, g1, z4, z4, z8, 0);
#else
  __builtin_amdgcn_tensor_load_to_lds(g0, g1, z4, z4, 0);
#endif
}
#endif

// ---- WMMA fragment loaders (CDNA5 16-bit A/B VGPR striping) ----
__device__ __forceinline__ v16h ldsA_frag(const _Float16* p, int row0, int k0, int stride) {
  const int lane = threadIdx.x & 31;
  const _Float16* r = p + (size_t)(row0 + (lane & 15)) * stride + k0 + (lane >> 4) * 8;
  Frag16 f;
  f.q[0] = *(const uint4*)(r);
  f.q[1] = *(const uint4*)(r + 16);
  return f.v;
}
__device__ __forceinline__ v16h ldsB_frag(const _Float16* p, int row0, int k0, int stride) {
  const int lane = threadIdx.x & 31;
  const _Float16* r = p + (size_t)(row0 + (lane & 15)) * stride + k0 + (lane >> 4) * 8;
  Frag16 f;
  f.q[0] = *(const uint4*)(r);
  f.q[1] = *(const uint4*)(r + 16);
  return f.v;
}

// ---------------- conversion kernels ----------------
__global__ void cvt_f32_to_f16_k(const float* __restrict__ src,
                                 _Float16* __restrict__ dst, int n) {
  int i = blockIdx.x * blockDim.x + threadIdx.x;
  if (i < n) dst[i] = (_Float16)src[i];
}

__global__ void cvt_transpose_f16_k(const float* __restrict__ src,
                                    _Float16* __restrict__ dst,
                                    int rows, int cols) {
  int i = blockIdx.x * blockDim.x + threadIdx.x;
  if (i < rows * cols) {
    int r = i / cols, c = i - r * cols;
    dst[(size_t)c * rows + r] = (_Float16)src[i];
  }
}

// ---------------- WMMA GEMM:  C[M][N] = A[M][K] * Bt[N][K]^T ----------------
// 256 threads = 8 wave32; block tile 256x128; each wave 64x64 = 4x4 WMMA accs.
// TDM path: double-buffered LDS panels, DMA overlapped with WMMA compute.
__launch_bounds__(256)
__global__ void gemm_f16_wmma_k(const _Float16* __restrict__ A,
                                const _Float16* __restrict__ Bt,
                                float* __restrict__ C,
                                int M, int N, int K) {
  const int m0   = blockIdx.y * 256;
  const int n0   = blockIdx.x * 128;
  const int tid  = threadIdx.x;
  const int lane = tid & 31;
  const int wave = tid >> 5;
  const int wm   = wave & 3;   // 4 row bands of 64
  const int wn   = wave >> 2;  // 2 col bands of 64
  const int l15  = lane & 15;
  const int half = lane >> 4;

  v8f acc[4][4] = {};

#if HAS_TDM
  __shared__ __align__(16) _Float16 sA[2][256 * 32];
  __shared__ __align__(16) _Float16 sB[2][128 * 32];

  if (wave == 0) {
    tdm_load_2d(A + (size_t)m0 * K, (unsigned)(size_t)&sA[0][0], 256, K);
    tdm_load_2d(Bt + (size_t)n0 * K, (unsigned)(size_t)&sB[0][0], 128, K);
  }
  int cur = 0;
  for (int kk = 0; kk < K; kk += 32) {
    if (wave == 0) __builtin_amdgcn_s_wait_tensorcnt((short)0);
    __syncthreads();                       // panel `cur` visible to all waves
    if (wave == 0 && kk + 32 < K) {        // DMA next panel while computing
      tdm_load_2d(A + (size_t)m0 * K + kk + 32, (unsigned)(size_t)&sA[cur ^ 1][0], 256, K);
      tdm_load_2d(Bt + (size_t)n0 * K + kk + 32, (unsigned)(size_t)&sB[cur ^ 1][0], 128, K);
    }
    v16h af[4], bf[4];
    #pragma unroll
    for (int mt = 0; mt < 4; ++mt) af[mt] = ldsA_frag(sA[cur], wm * 64 + mt * 16, 0, 32);
    #pragma unroll
    for (int nt = 0; nt < 4; ++nt) bf[nt] = ldsB_frag(sB[cur], wn * 64 + nt * 16, 0, 32);
    #pragma unroll
    for (int mt = 0; mt < 4; ++mt)
      #pragma unroll
      for (int nt = 0; nt < 4; ++nt)
        acc[mt][nt] = __builtin_amdgcn_wmma_f32_16x16x32_f16(
            false, af[mt], false, bf[nt], (short)0, acc[mt][nt], false, false);
    cur ^= 1;
  }
#else
  __shared__ __align__(16) _Float16 sA1[256 * 32];
  __shared__ __align__(16) _Float16 sB1[128 * 32];
  for (int kk = 0; kk < K; kk += 32) {
    #pragma unroll
    for (int e = tid; e < 1024; e += 256) {
      int row = e >> 2, seg = (e & 3) * 8;
      *(uint4*)&sA1[row * 32 + seg] = *(const uint4*)(A + (size_t)(m0 + row) * K + kk + seg);
    }
    #pragma unroll
    for (int e = tid; e < 512; e += 256) {
      int row = e >> 2, seg = (e & 3) * 8;
      *(uint4*)&sB1[row * 32 + seg] = *(const uint4*)(Bt + (size_t)(n0 + row) * K + kk + seg);
    }
    __syncthreads();
    v16h af[4], bf[4];
    #pragma unroll
    for (int mt = 0; mt < 4; ++mt) af[mt] = ldsA_frag(sA1, wm * 64 + mt * 16, 0, 32);
    #pragma unroll
    for (int nt = 0; nt < 4; ++nt) bf[nt] = ldsB_frag(sB1, wn * 64 + nt * 16, 0, 32);
    #pragma unroll
    for (int mt = 0; mt < 4; ++mt)
      #pragma unroll
      for (int nt = 0; nt < 4; ++nt)
        acc[mt][nt] = __builtin_amdgcn_wmma_f32_16x16x32_f16(
            false, af[mt], false, bf[nt], (short)0, acc[mt][nt], false, false);
    __syncthreads();
  }
#endif

  #pragma unroll
  for (int mt = 0; mt < 4; ++mt)
    #pragma unroll
    for (int nt = 0; nt < 4; ++nt) {
      int row = m0 + wm * 64 + mt * 16 + half * 8;
      int col = n0 + wn * 64 + nt * 16 + l15;
      #pragma unroll
      for (int r = 0; r < 8; ++r)
        C[(size_t)(row + r) * N + col] = acc[mt][nt][r];
    }
}

// ---------------- depthwise causal conv (K=4) + SiLU ----------------
__global__ void conv_silu_k(const float* __restrict__ x,
                            const float* __restrict__ w,
                            float* __restrict__ y) {
  long i = (long)blockIdx.x * blockDim.x + threadIdx.x;
  if (i >= (long)MTOK * DIM) return;
  int  d  = (int)(i % DIM);
  long bt = i / DIM;
  int  t  = (int)(bt % TT);
  float acc = 0.f;
  #pragma unroll
  for (int j = 0; j < KCONV; ++j) {
    int tt = t + j - (KCONV - 1);
    if (tt >= 0) acc += x[(bt + (j - (KCONV - 1))) * DIM + d] * w[d * KCONV + j];
  }
  y[i] = acc * (1.f / (1.f + __expf(-acc)));   // SiLU
}

// ---------------- per-head L2 norm (in place), optional extra scale ----------
__global__ void l2norm_head_k(float* __restrict__ x, float scale, int nvec) {
  int vIdx = blockIdx.x * blockDim.x + threadIdx.x;
  if (vIdx >= nvec) return;
  float* p = x + (size_t)vIdx * HD;
  float s = 1e-6f;
  #pragma unroll 8
  for (int j = 0; j < HD; ++j) s += p[j] * p[j];
  float r = rsqrtf(s) * scale;
  #pragma unroll 8
  for (int j = 0; j < HD; ++j) p[j] *= r;
}

// ---------------- beta = sigmoid(hs @ Wb), Wb [DIM][NH] ----------------
__global__ void beta_proj_k(const float* __restrict__ hs,
                            const float* __restrict__ Wb,
                            float* __restrict__ beta) {
  int i = blockIdx.x * blockDim.x + threadIdx.x;
  if (i >= MTOK * NH) return;
  int  h  = i & (NH - 1);
  long bt = i >> 4;
  const float* x = hs + bt * DIM;
  float s = 0.f;
  #pragma unroll 8
  for (int d = 0; d < DIM; ++d) s += x[d] * Wb[d * NH + h];
  beta[i] = 1.f / (1.f + __expf(-s));
}

// ---------------- chunked delta-rule recurrence (WMMA) ----------------
__launch_bounds__(256)
__global__ void delta_chunk_k(const float* __restrict__ q, const float* __restrict__ kq,
                              const float* __restrict__ v, const float* __restrict__ beta,
                              float* __restrict__ o) {
  extern __shared__ char smraw[];
  float*    S   = (float*)smraw;            // [128][128] f32 state
  float*    G   = S + HD * HD;              // [64][64]
  float*    R   = G + CC * CC;              // [64][128] residual / V-P
  float*    Ob  = R + CC * HD;              // [64][128] QS partial
  _Float16* Sh  = (_Float16*)(Ob + CC * HD);// [128][128] = S^T f16
  _Float16* Kh  = Sh + HD * HD;             // [64][128]
  _Float16* KhT = Kh + CC * HD;             // [128][64]
  _Float16* Qh  = KhT + HD * CC;            // [64][128]
  _Float16* YT  = Qh + CC * HD;             // [128][64]
  _Float16* QKm = YT + HD * CC;             // [64][64]
  float*    bl  = (float*)(QKm + CC * CC);  // [64]

  const int tid  = threadIdx.x;
  const int lane = tid & 31;
  const int wave = tid >> 5;
  const int l15  = lane & 15;
  const int half = lane >> 4;
  const int bh   = blockIdx.x;
  const int b    = bh >> 4, h = bh & (NH - 1);

  for (int e = tid; e < HD * HD; e += 256) S[e] = 0.f;
  __syncthreads();

  for (int ch = 0; ch < TT / CC; ++ch) {
    const long tg0 = (long)b * TT + (long)ch * CC;

    // phase A: stage chunk operands into LDS (f16, dual layouts)
    for (int e = tid; e < CC * HD; e += 256) {
      int t = e >> 7, d = e & (HD - 1);
      long g = (tg0 + t) * DIM + h * HD + d;
      _Float16 kv = (_Float16)kq[g];
      Kh[t * HD + d]  = kv;
      KhT[d * CC + t] = kv;
      Qh[t * HD + d]  = (_Float16)q[g];
    }
    for (int e = tid; e < HD * HD; e += 256) {
      int dk = e >> 7, d = e & (HD - 1);
      Sh[d * HD + dk] = (_Float16)S[dk * HD + d];
    }
    if (tid < CC) bl[tid] = beta[(tg0 + tid) * NH + h];
    __syncthreads();

    // phase C: P->R, G, QS->Ob, QK->QKm (96 WMMA tile-jobs over 8 waves)
    for (int job = wave; job < 96; job += 8) {
      int type, mt, nt;
      if (job < 32)      { type = 0; mt = job >> 3;        nt = job & 7;        }
      else if (job < 48) { type = 1; mt = (job - 32) >> 2; nt = (job - 32) & 3; }
      else if (job < 80) { type = 2; mt = (job - 48) >> 3; nt = (job - 48) & 7; }
      else               { type = 3; mt = (job - 80) >> 2; nt = (job - 80) & 3; }
      const _Float16* Ab = (type <= 1) ? Kh : Qh;
      const _Float16* Bb = (type == 0 || type == 2) ? Sh : Kh;
      v8f acc = {};
      #pragma unroll
      for (int kk2 = 0; kk2 < HD; kk2 += 32)
        acc = __builtin_amdgcn_wmma_f32_16x16x32_f16(
            false, ldsA_frag(Ab, mt * 16, kk2, HD),
            false, ldsB_frag(Bb, nt * 16, kk2, HD),
            (short)0, acc, false, false);
      #pragma unroll
      for (int rr = 0; rr < 8; ++rr) {
        int tr = mt * 16 + half * 8 + rr;
        int cI = nt * 16 + l15;
        if (type == 0) {
          long g = (tg0 + tr) * DIM + h * HD + cI;
          R[tr * HD + cI] = v[g] - acc[rr];
        } else if (type == 1) {
          G[tr * CC + cI] = acc[rr];
        } else if (type == 2) {
          Ob[tr * HD + cI] = acc[rr];
        } else {
          QKm[tr * CC + cI] = (cI <= tr) ? (_Float16)acc[rr] : (_Float16)0.f;
        }
      }
    }
    __syncthreads();

    // phase E: forward substitution; thread j owns column j (no syncs)
    if (tid < HD) {
      const int j = tid;
      for (int t = 0; t < CC; ++t) {
        float y = bl[t] * R[t * HD + j];
        YT[j * CC + t] = (_Float16)y;
        for (int s = t + 1; s < CC; ++s)
          R[s * HD + j] -= G[s * CC + t] * y;
      }
    }
    __syncthreads();

    // phase G: O = Ob + QKm@Y -> global ; S += K^T@Y
    for (int job = wave; job < 96; job += 8) {
      int type, mt, nt;
      if (job < 32) { type = 4; mt = job >> 3;        nt = job & 7; }
      else          { type = 5; mt = (job - 32) >> 3; nt = (job - 32) & 7; }
      v8f acc;
      #pragma unroll
      for (int rr = 0; rr < 8; ++rr) {
        int rI = mt * 16 + half * 8 + rr, cI = nt * 16 + l15;
        acc[rr] = (type == 4) ? Ob[rI * HD + cI] : S[rI * HD + cI];
      }
      const _Float16* Ab = (type == 4) ? QKm : KhT;
      #pragma unroll
      for (int kk2 = 0; kk2 < CC; kk2 += 32)
        acc = __builtin_amdgcn_wmma_f32_16x16x32_f16(
            false, ldsA_frag(Ab, mt * 16, kk2, CC),
            false, ldsB_frag(YT, nt * 16, kk2, CC),
            (short)0, acc, false, false);
      #pragma unroll
      for (int rr = 0; rr < 8; ++rr) {
        int rI = mt * 16 + half * 8 + rr, cI = nt * 16 + l15;
        if (type == 4) {
          long g = (tg0 + rI) * DIM + h * HD + cI;
          o[g] = acc[rr];
        } else {
          S[rI * HD + cI] = acc[rr];
        }
      }
    }
    __syncthreads();
  }
}

// ---------------- RMSNorm * norm_w * sigmoid(g) -> f16 ----------------
__global__ void post_gate_k(const float* __restrict__ o,
                            const float* __restrict__ g,
                            const float* __restrict__ nw,
                            _Float16* __restrict__ of16, int nvec) {
  int vIdx = blockIdx.x * blockDim.x + threadIdx.x;
  if (vIdx >= nvec) return;
  const float* p  = o + (size_t)vIdx * HD;
  const float* gp = g + (size_t)vIdx * HD;
  float s = 0.f;
  #pragma unroll 8
  for (int j = 0; j < HD; ++j) s += p[j] * p[j];
  float r = rsqrtf(s * (1.f / HD) + 1e-5f);
  #pragma unroll 8
  for (int j = 0; j < HD; ++j) {
    float gate = 1.f / (1.f + __expf(-gp[j]));
    of16[(size_t)vIdx * HD + j] = (_Float16)(p[j] * r * nw[j] * gate);
  }
}

// ============================ host launcher ============================
extern "C" void kernel_launch(void* const* d_in, const int* in_sizes, int n_in,
                              void* d_out, int out_size, void* d_ws, size_t ws_size,
                              hipStream_t stream) {
  const float* hs  = (const float*)d_in[0];
  const float* Wq  = (const float*)d_in[5];
  const float* Wk  = (const float*)d_in[6];
  const float* Wv  = (const float*)d_in[7];
  const float* Wo  = (const float*)d_in[8];
  const float* Wg  = (const float*)d_in[9];
  const float* Wb  = (const float*)d_in[10];
  const float* cwq = (const float*)d_in[11];
  const float* cwk = (const float*)d_in[12];
  const float* cwv = (const float*)d_in[13];
  const float* nw  = (const float*)d_in[14];
  float* out = (float*)d_out;

  char* ws = (char*)d_ws;
  size_t off = 0;
  auto carve = [&](size_t bytes) { char* p = ws + off; off += (bytes + 255) & ~(size_t)255; return p; };

  _Float16* hsf16 = (_Float16*)carve((size_t)MTOK * DIM * 2);
  _Float16* WqT   = (_Float16*)carve((size_t)DIM * DIM * 2);
  _Float16* WkT   = (_Float16*)carve((size_t)DIM * DIM * 2);
  _Float16* WvT   = (_Float16*)carve((size_t)DIM * DIM * 2);
  _Float16* WgT   = (_Float16*)carve((size_t)DIM * DIM * 2);
  _Float16* WoT   = (_Float16*)carve((size_t)DIM * DIM * 2);
  float* raw  = (float*)carve((size_t)MTOK * DIM * 4);
  float* qb   = (float*)carve((size_t)MTOK * DIM * 4);
  float* kb   = (float*)carve((size_t)MTOK * DIM * 4);
  float* vb   = (float*)carve((size_t)MTOK * DIM * 4);
  float* ob   = (float*)carve((size_t)MTOK * DIM * 4);
  float* beta = (float*)carve((size_t)MTOK * NH * 4);
  float* gb   = raw;
  _Float16* of16 = hsf16;

  const int nHS = MTOK * DIM;
  const int nW  = DIM * DIM;
  const int nV  = MTOK * NH;

  dim3 gGemm(DIM / 128, MTOK / 256);

  cvt_f32_to_f16_k<<<(nHS + 255) / 256, 256, 0, stream>>>(hs, hsf16, nHS);
  cvt_transpose_f16_k<<<(nW + 255) / 256, 256, 0, stream>>>(Wq, WqT, DIM, DIM);
  cvt_transpose_f16_k<<<(nW + 255) / 256, 256, 0, stream>>>(Wk, WkT, DIM, DIM);
  cvt_transpose_f16_k<<<(nW + 255) / 256, 256, 0, stream>>>(Wv, WvT, DIM, DIM);
  cvt_transpose_f16_k<<<(nW + 255) / 256, 256, 0, stream>>>(Wg, WgT, DIM, DIM);
  cvt_transpose_f16_k<<<(nW + 255) / 256, 256, 0, stream>>>(Wo, WoT, DIM, DIM);

  gemm_f16_wmma_k<<<gGemm, 256, 0, stream>>>(hsf16, WqT, raw, MTOK, DIM, DIM);
  conv_silu_k<<<(nHS + 255) / 256, 256, 0, stream>>>(raw, cwq, qb);
  gemm_f16_wmma_k<<<gGemm, 256, 0, stream>>>(hsf16, WkT, raw, MTOK, DIM, DIM);
  conv_silu_k<<<(nHS + 255) / 256, 256, 0, stream>>>(raw, cwk, kb);
  gemm_f16_wmma_k<<<gGemm, 256, 0, stream>>>(hsf16, WvT, raw, MTOK, DIM, DIM);
  conv_silu_k<<<(nHS + 255) / 256, 256, 0, stream>>>(raw, cwv, vb);
  gemm_f16_wmma_k<<<gGemm, 256, 0, stream>>>(hsf16, WgT, gb, MTOK, DIM, DIM);

  l2norm_head_k<<<(nV + 255) / 256, 256, 0, stream>>>(qb, 0.08838834764831845f, nV);
  l2norm_head_k<<<(nV + 255) / 256, 256, 0, stream>>>(kb, 1.0f, nV);
  beta_proj_k<<<(nV + 255) / 256, 256, 0, stream>>>(hs, Wb, beta);

  size_t deltaLds =
      (size_t)(HD * HD + CC * CC + CC * HD + CC * HD) * sizeof(float) +
      (size_t)(HD * HD + CC * HD + HD * CC + CC * HD + HD * CC + CC * CC) * sizeof(_Float16) +
      (size_t)CC * sizeof(float);
  delta_chunk_k<<<BB * NH, 256, deltaLds, stream>>>(qb, kb, vb, beta, ob);

  post_gate_k<<<(nV + 255) / 256, 256, 0, stream>>>(ob, gb, nw, of16, nV);
  gemm_f16_wmma_k<<<gGemm, 256, 0, stream>>>(of16, WoT, out, MTOK, DIM, DIM);
}
